// ELMO_20529943674941
// MI455X (gfx1250) — matrix operand order repaired and checked by reading, not compiled, not run on hardware
//
#include <hip/hip_runtime.h>
#include <hip/hip_bf16.h>

// ---------- CDNA5 WMMA fragment types ----------
typedef __attribute__((ext_vector_type(2)))  float  v2f;   // A/B of V_WMMA_F32_16X16X4_F32
typedef __attribute__((ext_vector_type(8)))  float  v8f;   // 16x16 f32 C/D
typedef __attribute__((ext_vector_type(16))) __bf16 v16bf; // A/B of V_WMMA_F32_16X16X32_BF16
// ---------- TDM descriptor vector types ----------
typedef __attribute__((ext_vector_type(4))) unsigned int v4u;
typedef __attribute__((ext_vector_type(8))) int          v8i;
typedef __attribute__((ext_vector_type(4))) int          v4i;

#define B_  16
#define S_  130
#define T_  129
#define E_  256
#define U_  512
#define G_  2048   // 4*U
#define V_  32000
#define P_  128    // output timesteps
#define M_  2048   // B_*P_ projection rows
#define KC_ 1024   // 2*U projection K

__device__ __forceinline__ float sigmoidf_(float x) {
  return 1.0f / (1.0f + __expf(-x));
}
__device__ __forceinline__ unsigned short f2bf_(float x) {
  unsigned int u = __float_as_uint(x);
  u += 0x7FFFu + ((u >> 16) & 1u);          // round to nearest even
  return (unsigned short)(u >> 16);
}

// ---------------------------------------------------------------------------
// 1) Embedding gather + mask
// ---------------------------------------------------------------------------
__global__ void elmo_embed_kernel(const int* __restrict__ seqs,
                                  const float* __restrict__ emb,
                                  float* __restrict__ X, float* __restrict__ mk) {
  int idx = blockIdx.x * blockDim.x + threadIdx.x;
  if (idx >= B_ * S_ * E_) return;
  int e  = idx % E_;
  int bs = idx / E_;
  int tok = seqs[bs];
  X[idx] = emb[(long)tok * E_ + e];
  if (e == 0) mk[bs] = (tok != 0) ? 1.0f : 0.0f;
}

// ---------------------------------------------------------------------------
// 2) Batched input projection: Z[t][b][n] = A_t[b,:] @ W + bias  (f32 WMMA)
// ---------------------------------------------------------------------------
__global__ void elmo_xk_wmma_kernel(const float* __restrict__ A,
                                    long long aTStride, long long aRowStride, int tmap,
                                    const float* __restrict__ W,
                                    const float* __restrict__ bias,
                                    float* __restrict__ Z, int K) {
  int wave = threadIdx.x >> 5;
  int lane = threadIdx.x & 31;
  int ntile = blockIdx.x * (blockDim.x >> 5) + wave;   // 0..127
  int t = blockIdx.y;
  int tidx = tmap ? (S_ - 1 - t) : t;

  int row = lane & 15;
  int hi  = lane >> 4;
  int col = lane & 15;
  int ncol = ntile * 16 + col;

  const float* arow = A + (long long)tidx * aTStride + (long long)row * aRowStride + 2 * hi;
  const float* wcol = W + (long long)(2 * hi) * G_ + ncol;

  v8f acc = {};
  for (int k = 0; k < K; k += 4) {
    v2f a; a.x = arow[k];                 a.y = arow[k + 1];
    v2f b; b.x = wcol[(long long)k * G_]; b.y = wcol[(long long)(k + 1) * G_];
    acc = __builtin_amdgcn_wmma_f32_16x16x4_f32(false, a, false, b,
                                                (short)0, acc, false, false);
  }
  float bs = bias[ncol];
  float* zt = Z + (long long)t * B_ * G_;
#pragma unroll
  for (int v = 0; v < 8; v++) {
    int r = v + 8 * hi;
    zt[(long long)r * G_ + ncol] = acc[v] + bs;
  }
}

// ---------------------------------------------------------------------------
// 3) Recurrent LSTM layer (both directions, grid.x = 2).
// ---------------------------------------------------------------------------
__global__ void __launch_bounds__(1024)
elmo_lstm_kernel(const float* __restrict__ Zf, const float* __restrict__ Zb,
                 const float* __restrict__ Rf, const float* __restrict__ Rb,
                 const float* __restrict__ mk,
                 float* __restrict__ Hf, float* __restrict__ Hb) {
  __shared__ float lds_h[U_ * B_];                        // [k][row], 32 KB

  int dir = blockIdx.x;
  const float* Z = dir ? Zb : Zf;
  const float* R = dir ? Rb : Rf;
  float*       H = dir ? Hb : Hf;

  int wave = threadIdx.x >> 5;
  int lane = threadIdx.x & 31;
  int row  = lane & 15;
  int hi   = lane >> 4;
  int ucol = wave * 16 + (lane & 15);

  for (int i = threadIdx.x; i < U_ * B_; i += blockDim.x) lds_h[i] = 0.0f;
  __syncthreads();

  float cst[8];
#pragma unroll
  for (int v = 0; v < 8; v++) cst[v] = 0.0f;

  const float* r0 = R + (long long)(2 * hi) * G_;

  for (int t = 0; t < T_; ++t) {
    const float* zt = Z + (long long)t * B_ * G_;
    v8f ai, af, ag, ao;
#pragma unroll
    for (int v = 0; v < 8; v++) {
      int r = v + 8 * hi;
      const float* zr = zt + (long long)r * G_;
      ai[v] = zr[0 * U_ + ucol];
      af[v] = zr[1 * U_ + ucol];
      ag[v] = zr[2 * U_ + ucol];
      ao[v] = zr[3 * U_ + ucol];
    }
    for (int k = 0; k < U_; k += 4) {
      v2f a;
      a.x = lds_h[(k + 2 * hi) * B_ + row];
      a.y = lds_h[(k + 2 * hi + 1) * B_ + row];
      const float* rk = r0 + (long long)k * G_;
      v2f b;
      b.x = rk[0 * U_ + ucol]; b.y = rk[G_ + 0 * U_ + ucol];
      ai = __builtin_amdgcn_wmma_f32_16x16x4_f32(false, a, false, b, (short)0, ai, false, false);
      b.x = rk[1 * U_ + ucol]; b.y = rk[G_ + 1 * U_ + ucol];
      af = __builtin_amdgcn_wmma_f32_16x16x4_f32(false, a, false, b, (short)0, af, false, false);
      b.x = rk[2 * U_ + ucol]; b.y = rk[G_ + 2 * U_ + ucol];
      ag = __builtin_amdgcn_wmma_f32_16x16x4_f32(false, a, false, b, (short)0, ag, false, false);
      b.x = rk[3 * U_ + ucol]; b.y = rk[G_ + 3 * U_ + ucol];
      ao = __builtin_amdgcn_wmma_f32_16x16x4_f32(false, a, false, b, (short)0, ao, false, false);
    }

    __syncthreads();
    int s = dir ? (S_ - 1 - t) : t;
    float* hOut = H + (long long)t * B_ * U_;
#pragma unroll
    for (int v = 0; v < 8; v++) {
      int r = v + 8 * hi;
      float m  = mk[r * S_ + s];
      float iv = sigmoidf_(ai[v]);
      float fv = sigmoidf_(af[v]);
      float gv = tanhf(ag[v]);
      float ov = sigmoidf_(ao[v]);
      float cn = fv * cst[v] + iv * gv;
      float hn = ov * tanhf(cn);
      float hprev = lds_h[ucol * B_ + r];
      float hfin = (m != 0.0f) ? hn : hprev;
      float cfin = (m != 0.0f) ? cn : cst[v];
      cst[v] = cfin;
      lds_h[ucol * B_ + r] = hfin;
      hOut[(long long)r * U_ + ucol] = hfin;
    }
    __syncthreads();
  }
}

// ---------------------------------------------------------------------------
// 4) Concat fwd/bwd hidden states -> bf16 rows [M_, 1024]
// ---------------------------------------------------------------------------
__global__ void elmo_hcat_kernel(const float* __restrict__ Hf1,
                                 const float* __restrict__ Hb1,
                                 unsigned short* __restrict__ Hc) {
  int idx = blockIdx.x * blockDim.x + threadIdx.x;
  if (idx >= M_ * KC_) return;
  int k = idx % KC_;
  int m = idx / KC_;
  int b = m / P_;
  int p = m % P_;
  float v;
  if (k < U_) v = Hf1[((long long)p * B_ + b) * U_ + k];
  else        v = Hb1[((long long)(P_ - 1 - p) * B_ + b) * U_ + (k - U_)];
  Hc[idx] = f2bf_(v);
}

// ---------------------------------------------------------------------------
// 5) Transpose + convert Wp[K,V] f32 -> Wt[V,K] bf16
// ---------------------------------------------------------------------------
__global__ void elmo_wt_kernel(const float* __restrict__ Wp,
                               unsigned short* __restrict__ Wt) {
  long long idx = (long long)blockIdx.x * blockDim.x + threadIdx.x;
  if (idx >= (long long)V_ * KC_) return;
  int k = (int)(idx % KC_);
  int n = (int)(idx / KC_);
  Wt[idx] = f2bf_(Wp[(long long)k * V_ + n]);
}

// ---------------------------------------------------------------------------
// 6) Projection GEMM: out[M_,V_] = Hc(bf16) @ WtT(bf16) + bp ; f32 accumulate.
//    A-tile (16 x 1024 bf16 = 32 KB) staged once per block into LDS by the
//    Tensor Data Mover (TENSOR_LOAD_TO_LDS + s_wait_tensorcnt), shared by all
//    8 waves; B fragments stream from L2 (Wt fits in the 192 MB L2).
// ---------------------------------------------------------------------------
union Frag16 { uint4 u[2]; v16bf v; };

__global__ void __launch_bounds__(256)
elmo_proj_wmma_kernel(const unsigned short* __restrict__ Hc,
                      const unsigned short* __restrict__ Wt,
                      const float* __restrict__ bp,
                      float* __restrict__ out) {
  __shared__ __align__(16) unsigned short ldsA[16 * KC_];   // 32 KB A tile

  int wave = threadIdx.x >> 5;
  int lane = threadIdx.x & 31;
  int ntile = blockIdx.x * (blockDim.x >> 5) + wave;      // 0..1999
  int mtile = blockIdx.y;                                 // 0..127
  int row = lane & 15;
  int hi  = lane >> 4;
  int col = lane & 15;
  int n = ntile * 16 + col;

  if (wave == 0) {
    // ---- TDM D# (CDNA5 ISA): 2-D tile, 16 rows x 1024 bf16
    unsigned long long ga =
        (unsigned long long)(uintptr_t)(Hc + (size_t)mtile * 16 * KC_);
    unsigned lds_off = (unsigned)(uintptr_t)(void*)ldsA;  // low 32 bits = LDS byte offset
    v4u g0;
    g0[0] = 1u;                                           // count=1, user descriptor
    g0[1] = lds_off;                                      // lds_addr
    g0[2] = (unsigned)ga;                                 // global_addr[31:0]
    g0[3] = (unsigned)((ga >> 32) & 0x1FFFFFFu) | (2u << 30); // addr[56:32] | type=2
    v8i g1;
    g1[0] = (int)(1u << 16);           // data_size=1 (2 bytes), mask=0, no pad/iterate
    g1[1] = (int)((unsigned)KC_ << 16);    // tensor_dim0[15:0] at bits 63:48 (=1024)
    g1[2] = (int)((unsigned)M_ << 16);     // tensor_dim0[31:16]=0 | tensor_dim1[15:0] (=2048)
    g1[3] = (int)((unsigned)KC_ << 16);    // tensor_dim1[31:16]=0 | tile_dim0 (=1024)
    g1[4] = 16;                        // tile_dim1=16, tile_dim2=0
    g1[5] = KC_;                       // tensor_dim0_stride[31:0] = 1024
    g1[6] = 0;                         // stride[47:32]=0, dim1_stride[15:0]=0
    g1[7] = 0;
    v4i g2 = {0, 0, 0, 0};             // 2-D: groups 2/3 unused (zero-filled)
    v4i g3 = {0, 0, 0, 0};
#if defined(__clang_major__) && (__clang_major__ >= 23)
    v8i g4 = {0, 0, 0, 0, 0, 0, 0, 0};
    __builtin_amdgcn_tensor_load_to_lds(g0, g1, g2, g3, g4, 0);
#else
    __builtin_amdgcn_tensor_load_to_lds(g0, g1, g2, g3, 0);
#endif
    __builtin_amdgcn_s_wait_tensorcnt(0);
  }
  __syncthreads();

  const unsigned short* arow = ldsA + row * KC_ + 8 * hi;          // ds_load_b128 path
  const unsigned short* brow = Wt + (long long)n * KC_ + 8 * hi;   // global

  v8f acc = {};
  for (int k = 0; k < KC_; k += 32) {
    __builtin_prefetch(brow + k + 128, 0, 1);             // global_prefetch_b8
    Frag16 A, Bm;
    A.u[0]  = *(const uint4*)(arow + k);                  // K = k+8hi   .. +7
    A.u[1]  = *(const uint4*)(arow + k + 16);             // K = k+16+8hi.. +7
    Bm.u[0] = *(const uint4*)(brow + k);                  // K = k+8hi   .. +7
    Bm.u[1] = *(const uint4*)(brow + k + 16);             // K = k+16+8hi.. +7
    acc = __builtin_amdgcn_wmma_f32_16x16x32_bf16(false, A.v, false, Bm.v,
                                                  (short)0, acc, false, false);
  }
  float bias = bp[n];
#pragma unroll
  for (int v = 0; v < 8; v++) {
    int r = mtile * 16 + v + 8 * hi;
    out[(long long)r * V_ + n] = acc[v] + bias;
  }
}

// ---------------------------------------------------------------------------
// Launcher
// ---------------------------------------------------------------------------
extern "C" void kernel_launch(void* const* d_in, const int* in_sizes, int n_in,
                              void* d_out, int out_size, void* d_ws, size_t ws_size,
                              hipStream_t stream) {
  const int*   seqs = (const int*)  d_in[0];
  const float* emb  = (const float*)d_in[1];
  const float* fk0  = (const float*)d_in[2];
  const float* fr0  = (const float*)d_in[3];
  const float* fb0  = (const float*)d_in[4];
  const float* fk1  = (const float*)d_in[5];
  const float* fr1  = (const float*)d_in[6];
  const float* fb1  = (const float*)d_in[7];
  const float* bk0  = (const float*)d_in[8];
  const float* br0  = (const float*)d_in[9];
  const float* bb0  = (const float*)d_in[10];
  const float* bk1  = (const float*)d_in[11];
  const float* br1  = (const float*)d_in[12];
  const float* bb1  = (const float*)d_in[13];
  const float* Wp   = (const float*)d_in[14];
  const float* bp   = (const float*)d_in[15];
  float* out = (float*)d_out;

  char* ws = (char*)d_ws;
  auto take = [&](size_t bytes) -> char* {
    char* p = ws;
    ws += (bytes + 255) & ~(size_t)255;
    return p;
  };
  float* X   = (float*)take((size_t)B_ * S_ * E_ * 4);
  float* Mk  = (float*)take((size_t)B_ * S_ * 4);
  float* Zf  = (float*)take((size_t)T_ * B_ * G_ * 4);
  float* Zb  = (float*)take((size_t)T_ * B_ * G_ * 4);
  float* Hf0 = (float*)take((size_t)T_ * B_ * U_ * 4);
  float* Hb0 = (float*)take((size_t)T_ * B_ * U_ * 4);
  float* Hf1 = (float*)take((size_t)T_ * B_ * U_ * 4);
  float* Hb1 = (float*)take((size_t)T_ * B_ * U_ * 4);
  unsigned short* Hc = (unsigned short*)take((size_t)M_ * KC_ * 2);
  unsigned short* Wt = (unsigned short*)take((size_t)V_ * KC_ * 2);

  {
    int n = B_ * S_ * E_;
    elmo_embed_kernel<<<(n + 255) / 256, 256, 0, stream>>>(seqs, emb, X, Mk);
  }
  {
    dim3 g(32, T_);
    elmo_xk_wmma_kernel<<<g, 128, 0, stream>>>(X, (long long)E_, (long long)S_ * E_, 0,
                                               fk0, fb0, Zf, E_);
    elmo_xk_wmma_kernel<<<g, 128, 0, stream>>>(X, (long long)E_, (long long)S_ * E_, 1,
                                               bk0, bb0, Zb, E_);
  }
  elmo_lstm_kernel<<<2, 1024, 0, stream>>>(Zf, Zb, fr0, br0, Mk, Hf0, Hb0);
  {
    dim3 g(32, T_);
    elmo_xk_wmma_kernel<<<g, 128, 0, stream>>>(Hf0, (long long)B_ * U_, (long long)U_, 0,
                                               fk1, fb1, Zf, U_);
    elmo_xk_wmma_kernel<<<g, 128, 0, stream>>>(Hb0, (long long)B_ * U_, (long long)U_, 0,
                                               bk1, bb1, Zb, U_);
  }
  elmo_lstm_kernel<<<2, 1024, 0, stream>>>(Zf, Zb, fr1, br1, Mk, Hf1, Hb1);
  {
    int n = M_ * KC_;
    elmo_hcat_kernel<<<(n + 255) / 256, 256, 0, stream>>>(Hf1, Hb1, Hc);
  }
  {
    long long n = (long long)V_ * KC_;
    elmo_wt_kernel<<<(unsigned)((n + 255) / 256), 256, 0, stream>>>(Wp, Wt);
  }
  {
    dim3 g(V_ / 16 / 8, M_ / 16);   // (250, 128), 8 waves/block
    elmo_proj_wmma_kernel<<<g, 256, 0, stream>>>(Hc, Wt, bp, out);
  }
}
